// LinearAttention_28475633172713
// MI455X (gfx1250) — matrix operand order repaired and checked by reading, not compiled, not run on hardware
//
#include <hip/hip_runtime.h>
#include <hip/hip_bf16.h>

typedef __attribute__((ext_vector_type(16))) __bf16 v16bf;
typedef __attribute__((ext_vector_type(8)))  __bf16 v8bf;
typedef __attribute__((ext_vector_type(4)))  __bf16 v4bf;
typedef __attribute__((ext_vector_type(8)))  float  v8f;

#define Bsz 8
#define Nsz 4096
#define Dsz 1024
#define LDK 40        // LDS row stride in bf16 (80B: 16B-aligned, low bank conflicts)
#define EPSF 1e-6f

// Load a 16-element bf16 fragment from an LDS row as two 16B chunks.
__device__ __forceinline__ v16bf ld_frag2(const __bf16* p, int o0, int o1) {
    const v8bf a = *(const v8bf*)(p + o0);
    const v8bf b = *(const v8bf*)(p + o1);
    v16bf r;
#pragma unroll
    for (int i = 0; i < 8; ++i) { r[i] = a[i]; r[i + 8] = b[i]; }
    return r;
}

__device__ __forceinline__ v8f wmma_bf16(v16bf a, v16bf b, v8f c) {
    return __builtin_amdgcn_wmma_f32_16x16x32_bf16(
        /*neg_a=*/false, a, /*neg_b=*/false, b,
        /*c_mod=*/(short)0, c, /*reuse_a=*/false, /*reuse_b=*/false);
}

// ---------------------------------------------------------------------------
// Kernel 1: projections.  C[n,e] = sum_d x[b,n,d] * W[e,d]  (+bias, relu+eps)
// grid = (E/128, N/128, B*3); p = z%3 selects (Q,K,V).
// Q stored [b][n][e] bf16; K,V stored transposed [b][e][n] bf16.
// ---------------------------------------------------------------------------
__global__ __launch_bounds__(256) void proj_kernel(
    const float* __restrict__ x,
    const float* __restrict__ Wq, const float* __restrict__ bq,
    const float* __restrict__ Wk, const float* __restrict__ bk,
    const float* __restrict__ Wv, const float* __restrict__ bv,
    __bf16* __restrict__ Q, __bf16* __restrict__ Kt, __bf16* __restrict__ Vt)
{
    __shared__ __bf16 As[128 * LDK];
    __shared__ __bf16 Bs[128 * LDK];

    const int tid  = threadIdx.x;
    const int lane = tid & 31;
    const int wave = tid >> 5;
    const int wr   = wave >> 1;      // 4 wave-rows of 32
    const int wc   = wave & 1;       // 2 wave-cols of 64
    const int kh   = lane >> 4;      // lane half selects K chunk
    const int lr   = lane & 15;

    const int p  = blockIdx.z % 3;
    const int b  = blockIdx.z / 3;
    const int e0 = blockIdx.x * 128;
    const int n0 = blockIdx.y * 128;

    const float* W    = (p == 0) ? Wq : (p == 1 ? Wk : Wv);
    const float* bias = (p == 0) ? bq : (p == 1 ? bk : bv);
    const float* xb   = x + (size_t)b * Nsz * Dsz;

    v8f acc[2][4];
#pragma unroll
    for (int i = 0; i < 2; ++i)
#pragma unroll
        for (int j = 0; j < 4; ++j)
            acc[i][j] = (v8f){0.f,0.f,0.f,0.f,0.f,0.f,0.f,0.f};

    for (int k0 = 0; k0 < Dsz; k0 += 32) {
        // Stage x tile (128 x 32 f32 -> bf16) and W tile (128 x 32) into LDS.
#pragma unroll
        for (int t = 0; t < 4; ++t) {
            int i = tid + t * 256;                 // 1024 float4-quads total
            int r = i >> 3, c = (i & 7) << 2;
            float4 v = *(const float4*)(xb + (size_t)(n0 + r) * Dsz + k0 + c);
            v4bf q; q[0] = (__bf16)v.x; q[1] = (__bf16)v.y;
                    q[2] = (__bf16)v.z; q[3] = (__bf16)v.w;
            *(v4bf*)&As[r * LDK + c] = q;
            float4 w = *(const float4*)(W + (size_t)(e0 + r) * Dsz + k0 + c);
            v4bf qw; qw[0] = (__bf16)w.x; qw[1] = (__bf16)w.y;
                     qw[2] = (__bf16)w.z; qw[3] = (__bf16)w.w;
            *(v4bf*)&Bs[r * LDK + c] = qw;
        }
        if (k0 + 32 < Dsz) { // prefetch next x tile (global_prefetch_b8)
            __builtin_prefetch(xb + (size_t)(n0 + (tid >> 1)) * Dsz + k0 + 32 + (tid & 1) * 16, 0, 1);
        }
        __syncthreads();

        v16bf af[2], bf[4];
#pragma unroll
        for (int i = 0; i < 2; ++i)
            af[i] = ld_frag2(&As[(wr * 32 + i * 16 + lr) * LDK], kh * 8, 16 + kh * 8);
#pragma unroll
        for (int j = 0; j < 4; ++j)
            bf[j] = ld_frag2(&Bs[(wc * 64 + j * 16 + lr) * LDK], kh * 8, 16 + kh * 8);
#pragma unroll
        for (int i = 0; i < 2; ++i)
#pragma unroll
            for (int j = 0; j < 4; ++j)
                acc[i][j] = wmma_bf16(af[i], bf[j], acc[i][j]);
        __syncthreads();
    }

    // Epilogue: C layout -> lane holds column (lane&15), rows r + 8*(lane>>4).
#pragma unroll
    for (int i = 0; i < 2; ++i) {
#pragma unroll
        for (int j = 0; j < 4; ++j) {
            const int col   = e0 + wc * 64 + j * 16 + lr;     // output feature e
            const int rbase = n0 + wr * 32 + i * 16 + 8 * kh; // token n
            const float bv_ = bias[col];
            float f[8];
#pragma unroll
            for (int r = 0; r < 8; ++r) {
                float v = acc[i][j][r] + bv_;
                if (p < 2) v = fmaxf(v, 0.f) + EPSF;          // feature map
                f[r] = v;
            }
            if (p == 0) {                                     // Q: [b][n][e]
                __bf16* Qb = Q + (size_t)b * Nsz * Dsz;
#pragma unroll
                for (int r = 0; r < 8; ++r)
                    Qb[(size_t)(rbase + r) * Dsz + col] = (__bf16)f[r];
            } else {                                          // K/V: [b][e][n]
                __bf16* T = (p == 1 ? Kt : Vt) + (size_t)b * Dsz * Nsz;
                v8bf pk;
#pragma unroll
                for (int r = 0; r < 8; ++r) pk[r] = (__bf16)f[r];
                *(v8bf*)&T[(size_t)col * Nsz + rbase] = pk;   // 16B packed store
            }
        }
    }
}

// ---------------------------------------------------------------------------
// Kernel 2: Ksum[b][d] = sum_n Kt[b][d][n]  (wave-per-row reduction)
// ---------------------------------------------------------------------------
__global__ __launch_bounds__(256) void ksum_kernel(const __bf16* __restrict__ Kt,
                                                   float* __restrict__ Ksum)
{
    const int row  = blockIdx.x * 8 + (threadIdx.x >> 5);   // b*D + d
    const int lane = threadIdx.x & 31;
    const __bf16* p = Kt + (size_t)row * Nsz;
    float s = 0.f;
    for (int i = lane; i < Nsz; i += 32) s += (float)p[i];
#pragma unroll
    for (int m = 16; m; m >>= 1) s += __shfl_xor(s, m, 32);
    if (lane == 0) Ksum[row] = s;
}

// ---------------------------------------------------------------------------
// Kernel 3: KV GEMM.  KV[d][e] = sum_n Kt[b][d][n] * Vt[b][e][n]
// A rows = Kt rows (contiguous n), B rows = Vt rows (contiguous n).
// Output stored transposed: KVt[b][e][d] bf16 (packed 16B stores).
// grid = (E/128, D/128, B)
// ---------------------------------------------------------------------------
__global__ __launch_bounds__(256) void kv_kernel(const __bf16* __restrict__ Kt,
                                                 const __bf16* __restrict__ Vt,
                                                 __bf16* __restrict__ KVt)
{
    __shared__ __bf16 As[128 * LDK];
    __shared__ __bf16 Bs[128 * LDK];

    const int tid  = threadIdx.x;
    const int lane = tid & 31;
    const int wave = tid >> 5;
    const int wr = wave >> 1, wc = wave & 1;
    const int kh = lane >> 4, lr = lane & 15;

    const int b  = blockIdx.z;
    const int e0 = blockIdx.x * 128;   // N dim of MMA
    const int d0 = blockIdx.y * 128;   // M dim of MMA
    const __bf16* A  = Kt + (size_t)b * Dsz * Nsz;
    const __bf16* Bm = Vt + (size_t)b * Dsz * Nsz;

    v8f acc[2][4];
#pragma unroll
    for (int i = 0; i < 2; ++i)
#pragma unroll
        for (int j = 0; j < 4; ++j)
            acc[i][j] = (v8f){0.f,0.f,0.f,0.f,0.f,0.f,0.f,0.f};

    for (int k0 = 0; k0 < Nsz; k0 += 32) {
#pragma unroll
        for (int t = 0; t < 2; ++t) {
            int i = tid + t * 256;                 // 512 8-elem chunks
            int r = i >> 2, c = (i & 3) << 3;
            *(v8bf*)&As[r * LDK + c] = *(const v8bf*)(A  + (size_t)(d0 + r) * Nsz + k0 + c);
            *(v8bf*)&Bs[r * LDK + c] = *(const v8bf*)(Bm + (size_t)(e0 + r) * Nsz + k0 + c);
        }
        if (k0 + 32 < Nsz)
            __builtin_prefetch(A + (size_t)(d0 + (tid >> 1)) * Nsz + k0 + 32, 0, 1);
        __syncthreads();

        v16bf af[2], bf[4];
#pragma unroll
        for (int i = 0; i < 2; ++i)
            af[i] = ld_frag2(&As[(wr * 32 + i * 16 + lr) * LDK], kh * 8, 16 + kh * 8);
#pragma unroll
        for (int j = 0; j < 4; ++j)
            bf[j] = ld_frag2(&Bs[(wc * 64 + j * 16 + lr) * LDK], kh * 8, 16 + kh * 8);
#pragma unroll
        for (int i = 0; i < 2; ++i)
#pragma unroll
            for (int j = 0; j < 4; ++j)
                acc[i][j] = wmma_bf16(af[i], bf[j], acc[i][j]);
        __syncthreads();
    }

    __bf16* KVb = KVt + (size_t)b * Dsz * Dsz;
#pragma unroll
    for (int i = 0; i < 2; ++i) {
#pragma unroll
        for (int j = 0; j < 4; ++j) {
            const int e     = e0 + wc * 64 + j * 16 + lr;
            const int dbase = d0 + wr * 32 + i * 16 + 8 * kh;
            v8bf pk;
#pragma unroll
            for (int r = 0; r < 8; ++r) pk[r] = (__bf16)acc[i][j][r];
            *(v8bf*)&KVb[(size_t)e * Dsz + dbase] = pk;       // KVt[e][d]
        }
    }
}

// ---------------------------------------------------------------------------
// Kernel 4: Z[b][n] = 1 / (sum_d Q[b][n][d] * Ksum[b][d] + eps)
// ---------------------------------------------------------------------------
__global__ __launch_bounds__(256) void z_kernel(const __bf16* __restrict__ Q,
                                                const float* __restrict__ Ksum,
                                                float* __restrict__ Z)
{
    const int row  = blockIdx.x * 8 + (threadIdx.x >> 5);   // b*N + n
    const int lane = threadIdx.x & 31;
    const __bf16* q = Q + (size_t)row * Dsz;
    const float* ks = Ksum + (size_t)(row >> 12) * Dsz;     // b = row / N
    float s = 0.f;
    for (int i = lane; i < Dsz; i += 32) s += (float)q[i] * ks[i];
#pragma unroll
    for (int m = 16; m; m >>= 1) s += __shfl_xor(s, m, 32);
    if (lane == 0) Z[row] = 1.f / (s + EPSF);
}

// ---------------------------------------------------------------------------
// Kernel 5: out[b][n][e] = Z[b][n] * sum_d Q[b][n][d] * KVt[b][e][d]   (f32)
// grid = (E/128, N/128, B)
// ---------------------------------------------------------------------------
__global__ __launch_bounds__(256) void out_kernel(const __bf16* __restrict__ Q,
                                                  const __bf16* __restrict__ KVt,
                                                  const float* __restrict__ Z,
                                                  float* __restrict__ out)
{
    __shared__ __bf16 As[128 * LDK];
    __shared__ __bf16 Bs[128 * LDK];

    const int tid  = threadIdx.x;
    const int lane = tid & 31;
    const int wave = tid >> 5;
    const int wr = wave >> 1, wc = wave & 1;
    const int kh = lane >> 4, lr = lane & 15;

    const int b  = blockIdx.z;
    const int e0 = blockIdx.x * 128;
    const int n0 = blockIdx.y * 128;
    const __bf16* A  = Q   + (size_t)b * Nsz * Dsz;   // rows n, ld D
    const __bf16* Bm = KVt + (size_t)b * Dsz * Dsz;   // rows e, ld D

    v8f acc[2][4];
#pragma unroll
    for (int i = 0; i < 2; ++i)
#pragma unroll
        for (int j = 0; j < 4; ++j)
            acc[i][j] = (v8f){0.f,0.f,0.f,0.f,0.f,0.f,0.f,0.f};

    for (int k0 = 0; k0 < Dsz; k0 += 32) {
#pragma unroll
        for (int t = 0; t < 2; ++t) {
            int i = tid + t * 256;
            int r = i >> 2, c = (i & 3) << 3;
            *(v8bf*)&As[r * LDK + c] = *(const v8bf*)(A  + (size_t)(n0 + r) * Dsz + k0 + c);
            *(v8bf*)&Bs[r * LDK + c] = *(const v8bf*)(Bm + (size_t)(e0 + r) * Dsz + k0 + c);
        }
        if (k0 + 32 < Dsz)
            __builtin_prefetch(A + (size_t)(n0 + (tid >> 1)) * Dsz + k0 + 32, 0, 1);
        __syncthreads();

        v16bf af[2], bf[4];
#pragma unroll
        for (int i = 0; i < 2; ++i)
            af[i] = ld_frag2(&As[(wr * 32 + i * 16 + lr) * LDK], kh * 8, 16 + kh * 8);
#pragma unroll
        for (int j = 0; j < 4; ++j)
            bf[j] = ld_frag2(&Bs[(wc * 64 + j * 16 + lr) * LDK], kh * 8, 16 + kh * 8);
#pragma unroll
        for (int i = 0; i < 2; ++i)
#pragma unroll
            for (int j = 0; j < 4; ++j)
                acc[i][j] = wmma_bf16(af[i], bf[j], acc[i][j]);
        __syncthreads();
    }

    float* ob = out + (size_t)b * Nsz * Dsz;
    const float* Zb = Z + (size_t)b * Nsz;
#pragma unroll
    for (int i = 0; i < 2; ++i) {
#pragma unroll
        for (int j = 0; j < 4; ++j) {
            const int col   = e0 + wc * 64 + j * 16 + lr;
            const int rbase = n0 + wr * 32 + i * 16 + 8 * kh;
#pragma unroll
            for (int r = 0; r < 8; ++r)
                ob[(size_t)(rbase + r) * Dsz + col] = acc[i][j][r] * Zb[rbase + r];
        }
    }
}

// ---------------------------------------------------------------------------
extern "C" void kernel_launch(void* const* d_in, const int* in_sizes, int n_in,
                              void* d_out, int out_size, void* d_ws, size_t ws_size,
                              hipStream_t stream) {
    const float* x  = (const float*)d_in[0];
    const float* Wq = (const float*)d_in[1];
    const float* bq = (const float*)d_in[2];
    const float* Wk = (const float*)d_in[3];
    const float* bk = (const float*)d_in[4];
    const float* Wv = (const float*)d_in[5];
    const float* bv = (const float*)d_in[6];
    float* out = (float*)d_out;

    char* w = (char*)d_ws;
    const size_t szQKV = (size_t)Bsz * Nsz * Dsz * 2;   // bf16
    __bf16* Q    = (__bf16*)w;  w += szQKV;
    __bf16* Kt   = (__bf16*)w;  w += szQKV;
    __bf16* Vt   = (__bf16*)w;  w += szQKV;
    __bf16* KVt  = (__bf16*)w;  w += (size_t)Bsz * Dsz * Dsz * 2;
    float*  Ksum = (float*)w;   w += (size_t)Bsz * Dsz * 4;
    float*  Z    = (float*)w;

    // 1) projections (Q,K,V via grid.z)
    proj_kernel<<<dim3(Dsz / 128, Nsz / 128, Bsz * 3), 256, 0, stream>>>(
        x, Wq, bq, Wk, bk, Wv, bv, Q, Kt, Vt);
    // 2) Ksum
    ksum_kernel<<<dim3((Bsz * Dsz) / 8), 256, 0, stream>>>(Kt, Ksum);
    // 3) KV = K^T V (per batch), stored transposed
    kv_kernel<<<dim3(Dsz / 128, Dsz / 128, Bsz), 256, 0, stream>>>(Kt, Vt, KVt);
    // 4) Z normalizer
    z_kernel<<<dim3((Bsz * Nsz) / 8), 256, 0, stream>>>(Q, Ksum, Z);
    // 5) out = Z * (Q @ KV)
    out_kernel<<<dim3(Dsz / 128, Nsz / 128, Bsz), 256, 0, stream>>>(Q, KVt, Z, out);
}
